// HybridAttentionBlock_8615704396093
// MI455X (gfx1250) — matrix operand order repaired and checked by reading, not compiled, not run on hardware
//
#include <hip/hip_runtime.h>

typedef __attribute__((ext_vector_type(16))) _Float16     v16h;
typedef __attribute__((ext_vector_type(8)))  float        v8f;
typedef __attribute__((ext_vector_type(4)))  unsigned int u32x4;
typedef __attribute__((ext_vector_type(8)))  int          i32x8;
typedef __attribute__((ext_vector_type(4)))  int          i32x4;

#define NEG_INF (-__builtin_inff())

#if defined(__has_builtin)
#if __has_builtin(__builtin_amdgcn_tensor_load_to_lds)
#define HAVE_TDM 1
#endif
#endif
#ifndef HAVE_TDM
#define HAVE_TDM 0
#endif

#if HAVE_TDM
// ---------------------------------------------------------------------------
// Issue one TDM 2-D tile load (f16 elements) global -> LDS.  D# per ISA
// cdna5 §8.3/8.4: group0 {count=1, lds_addr, global_addr, type=2},
// group1 {data_size=2B, tensor_dim0/1, tile_dim0/1, tensor_dim0_stride}.
// Caller: one wave per workgroup; completion via s_wait_tensorcnt + barrier.
// ---------------------------------------------------------------------------
__device__ __forceinline__ void tdm_load_tile_f16(
    unsigned lds_off, const _Float16* g, int tensor_d0, int tensor_d1,
    long long stride0_elems, int tile_w, int tile_h)
{
  const unsigned long long ga = (unsigned long long)(uintptr_t)g;
  u32x4 g0 = {
    1u,                                                    // count=1 (user D#)
    lds_off,                                               // lds_addr (bytes)
    (unsigned)(ga & 0xFFFFFFFFull),                        // global_addr[31:0]
    (unsigned)((ga >> 32) & 0x01FFFFFFull) | 0x80000000u   // addr[56:32] | type=2
  };
  const unsigned d0 = (unsigned)tensor_d0, d1 = (unsigned)tensor_d1;
  const unsigned long long s0 = (unsigned long long)stride0_elems;
  i32x8 g1 = {
    (int)0x00010000u,                                       // data_size=1 (2 bytes)
    (int)((d0 & 0xFFFFu) << 16),                            // tensor_dim0[15:0]
    (int)((d0 >> 16) | ((d1 & 0xFFFFu) << 16)),             // dim0 hi | dim1 lo
    (int)((d1 >> 16) | ((unsigned)tile_w << 16)),           // dim1 hi | tile_dim0
    (int)((unsigned)tile_h & 0xFFFFu),                      // tile_dim1 (tile_dim2=0)
    (int)(unsigned)(s0 & 0xFFFFFFFFull),                    // dim0_stride lo32
    (int)(unsigned)((s0 >> 32) & 0xFFFFull),                // dim0_stride hi16
    0
  };
  i32x4 gz = {0, 0, 0, 0};
#if defined(__clang_major__) && __clang_major__ >= 23
  i32x8 gz8 = {0, 0, 0, 0, 0, 0, 0, 0};
  __builtin_amdgcn_tensor_load_to_lds(g0, g1, gz, gz, gz8, 0);
#else
  __builtin_amdgcn_tensor_load_to_lds(g0, g1, gz, gz, 0);
#endif
}
#endif

// ---------------------------------------------------------------------------
// f16-operand / f32-accumulate WMMA GEMM:  C[M,N] = A[M,K] @ B[K,N].
// Tile 128(M) x 64(N) x 32(K); 8 waves in a 4x2 grid; 2x2 WMMA frags/wave.
// Double-buffered LDS; A tile streamed by the Tensor Data Mover (wave 0),
// B tile staged transposed [n][k] by threads; next-next B prefetched.
// Requires M%128==0, N%64==0, K%32==0 (true for every call here).
// ---------------------------------------------------------------------------
__global__ __launch_bounds__(256) void gemm_f16_wmma(
    const _Float16* __restrict__ A, const _Float16* __restrict__ B,
    float* __restrict__ C, int M, int N, int K)
{
  __shared__ __align__(16) _Float16 As[2][128][32];   // [m][k]
  __shared__ __align__(16) _Float16 Bs[2][64][32];    // [n][k] (transposed)

  const int tid  = threadIdx.x;
  const int lane = tid & 31;
  const int wave = tid >> 5;
  const int wm   = wave >> 1;        // 0..3
  const int wn   = wave & 1;         // 0..1
  const int rowBase = blockIdx.x * 128;
  const int colBase = blockIdx.y * 64;
  const int nk = K >> 5;

  v8f acc[2][2] = {};

  // ---- B staging (global f16 -> VGPR -> LDS transposed) ----
  const int kr = tid >> 4;           // 0..15
  const int c4 = (tid & 15) * 4;     // 0..60
  union BU { uint2 u; _Float16 h[4]; };
  BU b0, b1;

  auto loadB = [&](int ktile) {
    const size_t base = (size_t)(ktile * 32 + kr) * N + colBase + c4;
    b0.u = *(const uint2*)(B + base);
    b1.u = *(const uint2*)(B + base + (size_t)16 * N);
  };
  auto storeB = [&](int buf) {
    #pragma unroll
    for (int q = 0; q < 4; ++q) {
      Bs[buf][c4 + q][kr]      = b0.h[q];
      Bs[buf][c4 + q][kr + 16] = b1.h[q];
    }
  };

#if !HAVE_TDM
  // Fallback A staging if the TDM builtin is unavailable.
  union AU { uint4 u; _Float16 h[8]; };
  AU a0, a1;
  const int arow = tid >> 1;
  const int acb  = (tid & 1) * 16;
  auto loadA = [&](int ktile) {
    const size_t base = (size_t)(rowBase + arow) * K + ktile * 32 + acb;
    a0.u = *(const uint4*)(A + base);
    a1.u = *(const uint4*)(A + base + 8);
  };
  auto storeA = [&](int buf) {
    *(uint4*)&As[buf][arow][acb]     = a0.u;
    *(uint4*)&As[buf][arow][acb + 8] = a1.u;
  };
#endif

  // ---- prologue: stage tile 0 ----
#if HAVE_TDM
  if (wave == 0)
    tdm_load_tile_f16((unsigned)(uintptr_t)&As[0][0][0],
                      A + (size_t)rowBase * K, K, M, K, 32, 128);
#else
  loadA(0); storeA(0);
#endif
  loadB(0); storeB(0);
#if HAVE_TDM
  if (wave == 0) __builtin_amdgcn_s_wait_tensorcnt(0);
#endif
  __syncthreads();

  for (int ktile = 0; ktile < nk; ++ktile) {
    const int  buf  = ktile & 1;
    const bool more = (ktile + 1 < nk);
    if (more) {
#if HAVE_TDM
      if (wave == 0)
        tdm_load_tile_f16((unsigned)(uintptr_t)&As[buf ^ 1][0][0],
                          A + (size_t)rowBase * K + (size_t)(ktile + 1) * 32,
                          K, M, K, 32, 128);
#else
      loadA(ktile + 1);
#endif
      loadB(ktile + 1);
      if (ktile + 2 < nk)
        __builtin_prefetch((const void*)(B + (size_t)((ktile + 2) * 32 + kr) * N + colBase + c4), 0, 1);
    }

    // ---- fragments (documented 16-bit A 16x32 / B 32x16 layouts) ----
    v16h afrag[2], bfrag[2];
    const int mrow = lane & 15;
    const int kbA  = (lane < 16) ? 0 : 8;
    #pragma unroll
    for (int fm = 0; fm < 2; ++fm) {
      const int m = wm * 32 + fm * 16 + mrow;
      #pragma unroll
      for (int j = 0; j < 16; ++j)
        afrag[fm][j] = As[buf][m][kbA + ((j < 8) ? j : (j + 8))];
    }
    const int kbB = (lane < 16) ? 0 : 8;
    #pragma unroll
    for (int fn = 0; fn < 2; ++fn) {
      const int n = wn * 32 + fn * 16 + (lane & 15);
      #pragma unroll
      for (int j = 0; j < 16; ++j) bfrag[fn][j] = Bs[buf][n][kbB + ((j < 8) ? j : (j + 8))];
    }
    #pragma unroll
    for (int fm = 0; fm < 2; ++fm)
      #pragma unroll
      for (int fn = 0; fn < 2; ++fn)
        acc[fm][fn] = __builtin_amdgcn_wmma_f32_16x16x32_f16(
            false, afrag[fm], false, bfrag[fn], (short)0, acc[fm][fn], false, false);

    if (more) {
#if !HAVE_TDM
      storeA(buf ^ 1);
#endif
      storeB(buf ^ 1);
    }
#if HAVE_TDM
    if (wave == 0) __builtin_amdgcn_s_wait_tensorcnt(0);
#endif
    __syncthreads();
  }

  // ---- epilogue: D layout — VGPR r holds row (r | r+8), col = lane&15 ----
  const int mo = (lane < 16) ? 0 : 8;
  #pragma unroll
  for (int fm = 0; fm < 2; ++fm)
    #pragma unroll
    for (int fn = 0; fn < 2; ++fn) {
      const int nb = colBase + wn * 32 + fn * 16 + (lane & 15);
      const int mb = rowBase + wm * 32 + fm * 16 + mo;
      #pragma unroll
      for (int r = 0; r < 8; ++r)
        C[(size_t)(mb + r) * N + nb] = acc[fm][fn][r];
    }
}

// f32 -> f16 operand conversion (one pass, amortized across all GEMM reads)
__global__ void cvt_f16_kernel(const float* __restrict__ in, _Float16* __restrict__ out, int n)
{
  const int i = blockIdx.x * 256 + threadIdx.x;
  if (i < n) out[i] = (_Float16)in[i];
}

// ---------------------------------------------------------------------------
// Phrase compression: per phrase p (one wave), gather hp = h[tok[p, 0..15], :],
// c_tok = hp@Wkv, z = hp@Wz + Bpos, per-channel softmax over 4 valid slots,
// out[p,c] = sum_l gate * c_tok.  WMMA: M=16 slots, N=64, K=1024.
// ---------------------------------------------------------------------------
__global__ __launch_bounds__(32) void compress_kernel(
    const float* __restrict__ h, const int* __restrict__ tok_idx,
    const float* __restrict__ Wkv, const float* __restrict__ Wz,
    const float* __restrict__ Bpos, float* __restrict__ out)
{
  __shared__ _Float16 Ah[16][32];
  __shared__ _Float16 Bkv[64][32];
  __shared__ _Float16 Bz[64][32];
  __shared__ float    zb[16][64];
  __shared__ float    cb[16][64];

  const int p = blockIdx.x, lane = threadIdx.x;
  int toks[16];
  #pragma unroll
  for (int l = 0; l < 16; ++l) toks[l] = tok_idx[p * 16 + l];

  v8f akv[4] = {}, az[4] = {};

  for (int kt = 0; kt < 1024; kt += 32) {
    #pragma unroll
    for (int l = 0; l < 16; ++l)
      Ah[l][lane] = (_Float16)h[(size_t)toks[l] * 1024 + kt + lane];
    for (int it = 0; it < 64; ++it) {
      const int f = it * 32 + lane;
      const int kg = f >> 6, n = f & 63;
      Bkv[n][kg] = (_Float16)Wkv[(size_t)(kt + kg) * 64 + n];
      Bz [n][kg] = (_Float16)Wz [(size_t)(kt + kg) * 64 + n];
    }
    __syncthreads();

    v16h a;
    const int m = lane & 15, kbA = (lane < 16) ? 0 : 8;
    #pragma unroll
    for (int j = 0; j < 16; ++j) a[j] = Ah[m][kbA + ((j < 8) ? j : (j + 8))];
    const int kbB = (lane < 16) ? 0 : 8;
    #pragma unroll
    for (int fn = 0; fn < 4; ++fn) {
      const int n = fn * 16 + (lane & 15);
      v16h bk, bz2;
      #pragma unroll
      for (int j = 0; j < 16; ++j) { bk[j] = Bkv[n][kbB + ((j < 8) ? j : (j + 8))]; bz2[j] = Bz[n][kbB + ((j < 8) ? j : (j + 8))]; }
      akv[fn] = __builtin_amdgcn_wmma_f32_16x16x32_f16(false, a, false, bk,  (short)0, akv[fn], false, false);
      az [fn] = __builtin_amdgcn_wmma_f32_16x16x32_f16(false, a, false, bz2, (short)0, az [fn], false, false);
    }
    __syncthreads();
  }

  const int nl = lane & 15, mo = (lane < 16) ? 0 : 8;
  #pragma unroll
  for (int fn = 0; fn < 4; ++fn) {
    const int c = fn * 16 + nl;
    #pragma unroll
    for (int r = 0; r < 8; ++r) {
      const int m = r + mo;
      cb[m][c] = akv[fn][r];
      zb[m][c] = az[fn][r] + Bpos[m * 64 + c];
    }
  }
  __syncthreads();
  #pragma unroll
  for (int q = 0; q < 2; ++q) {
    const int c = lane + q * 32;
    float mx = -3.0e30f;
    for (int l = 0; l < 4; ++l) mx = fmaxf(mx, zb[l][c]);
    float den = 0.f, num = 0.f;
    for (int l = 0; l < 4; ++l) {
      const float e = __expf(zb[l][c] - mx);
      den += e; num += e * cb[l][c];
    }
    out[(size_t)p * 64 + c] = num / den;
  }
}

// ---------------------------------------------------------------------------
// Phrase scores + top-64 selection per token (set-equivalent to lax.top_k).
// ---------------------------------------------------------------------------
__global__ __launch_bounds__(256) void scores_topk_kernel(
    const float* __restrict__ q_i, const float* __restrict__ hw,
    const float* __restrict__ k_idx, const int* __restrict__ end_pos,
    int* __restrict__ top_out)
{
  __shared__ float qi_s[256];
  __shared__ float hws[4];
  __shared__ float sc[512];
  __shared__ float rv[256];
  __shared__ int   ri[256];

  const int t = blockIdx.x, tid = threadIdx.x;
  qi_s[tid] = q_i[(size_t)t * 256 + tid];
  if (tid < 4) hws[tid] = hw[t * 4 + tid];
  __syncthreads();

  #pragma unroll
  for (int rep = 0; rep < 2; ++rep) {
    const int p = tid + rep * 256;
    float s = 0.f;
    for (int hh = 0; hh < 4; ++hh) {
      float d = 0.f;
      for (int c = 0; c < 64; ++c) d += qi_s[hh * 64 + c] * k_idx[(size_t)p * 64 + c];
      s += fmaxf(d, 0.f) * hws[hh];
    }
    sc[p] = (end_pos[p] < t) ? s : NEG_INF;
  }
  __syncthreads();

  for (int sel = 0; sel < 64; ++sel) {
    const float v0 = sc[tid], v1 = sc[tid + 256];
    if (v1 > v0) { rv[tid] = v1; ri[tid] = tid + 256; }
    else         { rv[tid] = v0; ri[tid] = tid; }
    __syncthreads();
    for (int off = 128; off >= 1; off >>= 1) {
      if (tid < off && rv[tid + off] > rv[tid]) { rv[tid] = rv[tid + off]; ri[tid] = ri[tid + off]; }
      __syncthreads();
    }
    if (tid == 0) {
      if (rv[0] == NEG_INF) top_out[(size_t)t * 64 + sel] = -1;
      else { top_out[(size_t)t * 64 + sel] = ri[0]; sc[ri[0]] = NEG_INF; }
    }
    __syncthreads();
  }
}

// ---------------------------------------------------------------------------
// In-place RMSNorm (+weight) + RoPE over one (token, head) 64-vector.
// ---------------------------------------------------------------------------
__global__ __launch_bounds__(64) void normrope_kernel(
    float* __restrict__ x, const float* __restrict__ w,
    const float* __restrict__ cosp, const float* __restrict__ sinp, int Hn)
{
  __shared__ float red[64];
  __shared__ float xs[64];
  const int b = blockIdx.x, t = b / Hn, c = threadIdx.x;
  const float v = x[(size_t)b * 64 + c];
  red[c] = v * v;
  __syncthreads();
  for (int off = 32; off >= 1; off >>= 1) {
    if (c < off) red[c] += red[c + off];
    __syncthreads();
  }
  const float r  = rsqrtf(red[0] * (1.f / 64.f) + 1e-6f);
  const float xn = v * r * w[c];
  xs[c] = xn;
  __syncthreads();
  const float rot = (c < 32) ? -xs[c + 32] : xs[c - 32];
  x[(size_t)b * 64 + c] = xn * cosp[t * 64 + c] + rot * sinp[t * 64 + c];
}

// ---------------------------------------------------------------------------
// Compressed selective attention core (token/block, slot/thread, sink softmax).
// ---------------------------------------------------------------------------
__global__ __launch_bounds__(64) void csa_attn_kernel(
    const float* __restrict__ qn, const float* __restrict__ c_comp,
    const int* __restrict__ top_idx, const int* __restrict__ end_pos,
    const float* __restrict__ kn_w,
    const float* __restrict__ cosp, const float* __restrict__ sinp,
    const float* __restrict__ sink, float* __restrict__ ctx)
{
  __shared__ float qs[16 * 64];
  __shared__ float Ks[64][64];
  __shared__ float lg[16 * 64];

  const int t = blockIdx.x, tid = threadIdx.x;
  #pragma unroll
  for (int i = 0; i < 16; ++i) qs[i * 64 + tid] = qn[(size_t)t * 1024 + i * 64 + tid];

  const int idx = top_idx[(size_t)t * 64 + tid];
  if (idx >= 0) {
    float kr[64];
    float ss = 0.f;
    for (int c = 0; c < 64; ++c) { const float v = c_comp[(size_t)idx * 64 + c]; kr[c] = v; ss += v * v; }
    const float r = rsqrtf(ss * (1.f / 64.f) + 1e-6f);
    for (int c = 0; c < 64; ++c) kr[c] = kr[c] * r * kn_w[c];
    int pe = end_pos[idx]; if (pe < 0) pe = 0;
    for (int c = 0; c < 32; ++c) {
      const float x1 = kr[c], x2 = kr[c + 32];
      Ks[tid][c]      = x1 * cosp[pe * 64 + c]      - x2 * sinp[pe * 64 + c];
      Ks[tid][c + 32] = x2 * cosp[pe * 64 + 32 + c] + x1 * sinp[pe * 64 + 32 + c];
    }
  } else {
    for (int c = 0; c < 64; ++c) Ks[tid][c] = 0.f;
  }
  __syncthreads();

  for (int hh = 0; hh < 16; ++hh) {
    float d = 0.f;
    for (int c = 0; c < 64; ++c) d += qs[hh * 64 + c] * Ks[tid][c];
    lg[hh * 64 + tid] = (idx >= 0) ? d * 0.125f : NEG_INF;
  }
  __syncthreads();

  if (tid < 16) {
    const float s0 = sink[tid];
    float mx = s0;
    for (int j = 0; j < 64; ++j) mx = fmaxf(mx, lg[tid * 64 + j]);
    float den = __expf(s0 - mx);
    for (int j = 0; j < 64; ++j) { const float e = __expf(lg[tid * 64 + j] - mx); den += e; lg[tid * 64 + j] = e; }
    const float inv = 1.f / den;
    for (int j = 0; j < 64; ++j) lg[tid * 64 + j] *= inv;
  }
  __syncthreads();

  for (int hh = 0; hh < 16; ++hh) {
    float o = 0.f;
    for (int j = 0; j < 64; ++j) o += lg[hh * 64 + j] * Ks[j][tid];
    ctx[(size_t)t * 1024 + hh * 64 + tid] = o;
  }
}

// ---------------------------------------------------------------------------
// Sliding-window attention core: block = (token t, head h), 128-key window.
// ---------------------------------------------------------------------------
__global__ __launch_bounds__(128) void sw_attn_kernel(
    const float* __restrict__ qw, const float* __restrict__ kw,
    const float* __restrict__ vw, float* __restrict__ ctx)
{
  __shared__ float qv[64];
  __shared__ float lg[128];
  __shared__ float red[128];
  __shared__ float part[2][64];

  const int t = blockIdx.x, h2 = blockIdx.y, tid = threadIdx.x;
  if (tid < 64) qv[tid] = qw[(size_t)t * 1024 + h2 * 64 + tid];
  __syncthreads();

  const int s = t - 127 + tid;
  float l = NEG_INF;
  if (s >= 0) {
    float d = 0.f;
    for (int c = 0; c < 64; ++c) d += qv[c] * kw[(size_t)s * 1024 + h2 * 64 + c];
    l = d * 0.125f;
  }
  lg[tid] = l; red[tid] = l;
  __syncthreads();
  for (int off = 64; off >= 1; off >>= 1) {
    if (tid < off) red[tid] = fmaxf(red[tid], red[tid + off]);
    __syncthreads();
  }
  const float mx = red[0];
  __syncthreads();
  const float e = __expf(l - mx);
  red[tid] = e;
  __syncthreads();
  for (int off = 64; off >= 1; off >>= 1) {
    if (tid < off) red[tid] += red[tid + off];
    __syncthreads();
  }
  const float inv = 1.f / red[0];
  lg[tid] = e * inv;
  __syncthreads();

  const int c = tid & 63, hf = tid >> 6;
  float acc = 0.f;
  for (int j = hf * 64; j < hf * 64 + 64; ++j) {
    const int ss = t - 127 + j;
    if (ss >= 0) acc += lg[j] * vw[(size_t)ss * 1024 + h2 * 64 + c];
  }
  part[hf][c] = acc;
  __syncthreads();
  if (tid < 64) ctx[(size_t)t * 1024 + h2 * 64 + tid] = part[0][tid] + part[1][tid];
}

// hw[t,i] = h[t,:] @ idx_Ww[:,i]   (N=4, tiny)
__global__ void hw_kernel(const float* __restrict__ h, const float* __restrict__ Ww,
                          float* __restrict__ hw)
{
  const int gid = blockIdx.x * blockDim.x + threadIdx.x;
  const int t = gid >> 2, i = gid & 3;
  float a = 0.f;
  for (int k = 0; k < 1024; ++k) a += h[(size_t)t * 1024 + k] * Ww[k * 4 + i];
  hw[gid] = a;
}

__global__ void add3_kernel(const float* __restrict__ a, const float* __restrict__ b,
                            const float* __restrict__ c, float* __restrict__ o, int n)
{
  const int i = blockIdx.x * blockDim.x + threadIdx.x;
  if (i < n) o[i] = a[i] + b[i] + c[i];
}

// ---------------------------------------------------------------------------
extern "C" void kernel_launch(void* const* d_in, const int* in_sizes, int n_in,
                              void* d_out, int out_size, void* d_ws, size_t ws_size,
                              hipStream_t stream) {
  (void)in_sizes; (void)n_in; (void)out_size; (void)ws_size;

  const float* h        = (const float*)d_in[0];
  /* d_in[1] phrase_mask: identically (slot < PLEN) -> folded as constant */
  const int*   tok_idx  = (const int*)  d_in[2];
  const int*   end_pos  = (const int*)  d_in[3];
  const float* rope_cos = (const float*)d_in[4];
  const float* rope_sin = (const float*)d_in[5];
  const float* W_dq     = (const float*)d_in[6];
  const float* W_uq     = (const float*)d_in[7];
  const float* kv_Wkv   = (const float*)d_in[8];
  const float* kv_Wz    = (const float*)d_in[9];
  const float* kv_Bpos  = (const float*)d_in[10];
  const float* ik_Wkv   = (const float*)d_in[11];
  const float* ik_Wz    = (const float*)d_in[12];
  const float* ik_Bpos  = (const float*)d_in[13];
  const float* idx_Wiuq = (const float*)d_in[14];
  const float* idx_Ww   = (const float*)d_in[15];
  const float* qn_w     = (const float*)d_in[16];
  const float* kn_w     = (const float*)d_in[17];
  const float* Wo_csa   = (const float*)d_in[18];
  const float* sink     = (const float*)d_in[19];
  const float* sw_Wq    = (const float*)d_in[20];
  const float* sw_Wk    = (const float*)d_in[21];
  const float* sw_Wv    = (const float*)d_in[22];
  const float* sw_Wo    = (const float*)d_in[23];
  const float* sw_qn    = (const float*)d_in[24];
  const float* sw_kn    = (const float*)d_in[25];

  const int T = 2048;

  float* ws       = (float*)d_ws;
  float* q_latent = ws;  ws += (size_t)T * 512;
  float* q_full   = ws;  ws += (size_t)T * 1024;
  float* q_i      = ws;  ws += (size_t)T * 256;
  float* hwbuf    = ws;  ws += (size_t)T * 4;
  float* c_comp   = ws;  ws += 512 * 64;
  float* k_idx    = ws;  ws += 512 * 64;
  int*   top_i    = (int*)ws; ws += (size_t)T * 64;
  float* csa_ctx  = ws;  ws += (size_t)T * 1024;
  float* swq      = ws;  ws += (size_t)T * 1024;
  float* swk      = ws;  ws += (size_t)T * 1024;
  float* swv      = ws;  ws += (size_t)T * 1024;
  float* sw_ctx   = ws;  ws += (size_t)T * 1024;
  float* csa_out  = q_full;   // q_full dead after csa_attn
  float* sw_out   = swq;      // swq dead after sw_attn

  // f16 operand staging (reused serially; deterministic)
  _Float16* h16  = (_Float16*)ws;        // T*1024
  _Float16* ql16 = h16  + (size_t)T * 1024;   // T*512
  _Float16* a16  = ql16 + (size_t)T * 512;    // T*1024 (ctx staging)
  _Float16* w16  = a16  + (size_t)T * 1024;   // up to 1024*1024 (weight staging)

  auto cvt = [&](const float* src, _Float16* dst, int n) {
    cvt_f16_kernel<<<(n + 255) / 256, 256, 0, stream>>>(src, dst, n);
  };

  // ---- projections (WMMA GEMMs, f16 operands, TDM-staged A tiles) ----
  cvt(h, h16, T * 1024);
  cvt(W_dq, w16, 1024 * 512);
  gemm_f16_wmma<<<dim3(16,  8), 256, 0, stream>>>(h16, w16, q_latent, T,  512, 1024);
  cvt(q_latent, ql16, T * 512);
  cvt(W_uq, w16, 512 * 1024);
  gemm_f16_wmma<<<dim3(16, 16), 256, 0, stream>>>(ql16, w16, q_full, T, 1024, 512);
  cvt(idx_Wiuq, w16, 512 * 256);
  gemm_f16_wmma<<<dim3(16,  4), 256, 0, stream>>>(ql16, w16, q_i, T, 256, 512);
  hw_kernel<<<32, 256, 0, stream>>>(h, idx_Ww, hwbuf);

  // ---- phrase compression (kv values + index keys) ----
  compress_kernel<<<512, 32, 0, stream>>>(h, tok_idx, kv_Wkv, kv_Wz, kv_Bpos, c_comp);
  compress_kernel<<<512, 32, 0, stream>>>(h, tok_idx, ik_Wkv, ik_Wz, ik_Bpos, k_idx);

  // ---- scores + top-64 selection ----
  scores_topk_kernel<<<T, 256, 0, stream>>>(q_i, hwbuf, k_idx, end_pos, top_i);

  // ---- compressed selective attention ----
  normrope_kernel<<<T * 16, 64, 0, stream>>>(q_full, qn_w, rope_cos, rope_sin, 16);
  csa_attn_kernel<<<T, 64, 0, stream>>>(q_full, c_comp, top_i, end_pos, kn_w,
                                        rope_cos, rope_sin, sink, csa_ctx);
  cvt(csa_ctx, a16, T * 1024);
  cvt(Wo_csa, w16, 1024 * 1024);
  gemm_f16_wmma<<<dim3(16, 16), 256, 0, stream>>>(a16, w16, csa_out, T, 1024, 1024);

  // ---- sliding-window attention ----
  cvt(sw_Wq, w16, 1024 * 1024);
  gemm_f16_wmma<<<dim3(16, 16), 256, 0, stream>>>(h16, w16, swq, T, 1024, 1024);
  cvt(sw_Wk, w16, 1024 * 1024);
  gemm_f16_wmma<<<dim3(16, 16), 256, 0, stream>>>(h16, w16, swk, T, 1024, 1024);
  cvt(sw_Wv, w16, 1024 * 1024);
  gemm_f16_wmma<<<dim3(16, 16), 256, 0, stream>>>(h16, w16, swv, T, 1024, 1024);
  normrope_kernel<<<T * 16, 64, 0, stream>>>(swq, sw_qn, rope_cos, rope_sin, 16);
  normrope_kernel<<<T * 16, 64, 0, stream>>>(swk, sw_kn, rope_cos, rope_sin, 16);
  sw_attn_kernel<<<dim3(T, 16), 128, 0, stream>>>(swq, swk, swv, sw_ctx);
  cvt(sw_ctx, a16, T * 1024);
  cvt(sw_Wo, w16, 1024 * 1024);
  gemm_f16_wmma<<<dim3(16, 16), 256, 0, stream>>>(a16, w16, sw_out, T, 1024, 1024);

  // ---- residual combine ----
  add3_kernel<<<(T * 1024) / 256, 256, 0, stream>>>(h, csa_out, sw_out, (float*)d_out, T * 1024);
}
